// MultiHeadAttention_62423054680360
// MI455X (gfx1250) — compile-verified, matrix operand order checked by reading, not run on hardware
//
#include <hip/hip_runtime.h>
#include <hip/hip_bf16.h>
#include <math.h>

#define B_ 4
#define S_ 2048
#define D_ 512
#define H_ 8
#define E_ 64

typedef __attribute__((ext_vector_type(16))) __bf16 v16bf;
typedef __attribute__((ext_vector_type(8)))  __bf16 v8bf;
typedef __attribute__((ext_vector_type(8)))  float  v8f;
typedef __attribute__((ext_vector_type(4)))  unsigned int v4u;
typedef __attribute__((ext_vector_type(8)))  int v8i;
typedef __attribute__((ext_vector_type(4)))  int v4i;

#if defined(__has_builtin)
#  if __has_builtin(__builtin_amdgcn_tensor_load_to_lds)
#    define HAVE_TDM 1
#  else
#    define HAVE_TDM 0
#  endif
#else
#  define HAVE_TDM 0
#endif

__device__ __forceinline__ v16bf cat8(v8bf lo, v8bf hi){
  v16bf r;
#pragma unroll
  for(int i=0;i<8;++i){ r[i]=lo[i]; r[i+8]=hi[i]; }
  return r;
}

// Load a 16(rows) x 32(K) bf16 WMMA A/B operand from row-major memory.
// ISA layout (CDNA5 ISA): lanes 0-15: row=lane, K {0..7,16..23};
// lanes 16-31: row=lane-16, K {8..15,24..31}. Two 16B contiguous loads/lane.
__device__ __forceinline__ v16bf load_tile16x32(const __bf16* base, int ld, int lane){
  const __bf16* p = base + (size_t)(lane & 15) * ld + ((lane & 16) ? 8 : 0);
  v8bf lo = *(const v8bf*)(p);
  v8bf hi = *(const v8bf*)(p + 16);
  return cat8(lo, hi);
}

__device__ __forceinline__ v8f wmma_bf16(v16bf a, v16bf b, v8f c){
  return __builtin_amdgcn_wmma_f32_16x16x32_bf16(false, a, false, b, (short)0, c, false, false);
}

// One 16-byte lane-wise async DMA: global -> LDS (ASYNCcnt path, gfx1250).
// lds_off = low 32 bits of generic LDS address.
__device__ __forceinline__ void async_copy16(unsigned lds_off, const void* gptr){
  unsigned long long ga = (unsigned long long)(uintptr_t)gptr;
  asm volatile("global_load_async_to_lds_b128 %0, %1, off"
               :: "v"(lds_off), "v"(ga) : "memory");
}

// TDM: issue a 1-D contiguous tile load (nbytes <= 65535, here 4096) global->LDS.
// D# built per CDNA5 ISA (group0 + group1; groups 2/3 zero).
__device__ __forceinline__ void tdm_load_1d(unsigned lds_off, unsigned long long ga,
                                            unsigned nbytes){
#if HAVE_TDM
  v4u g0;
  g0[0] = 1u;                                        // count=1, user descriptor
  g0[1] = lds_off;                                   // lds_addr
  g0[2] = (unsigned)(ga & 0xFFFFFFFFu);              // global_addr[31:0]
  g0[3] = ((unsigned)(ga >> 32) & 0x01FFFFFFu)       // global_addr[56:32]
        | (2u << 30);                                // type=2 ("image")
  v8i g1;
  g1[0] = 0;                                         // mask=0, data_size=1B, no flags
  g1[1] = (int)(nbytes << 16);                       // tensor_dim0 lo16 (bits 79:48)
  g1[2] = (int)(1u << 16);                           // tensor_dim0 hi16=0, tensor_dim1=1
  g1[3] = (int)(nbytes << 16);                       // tile_dim0 (bits 127:112)
  g1[4] = 0;                                         // tile_dim1=0 (1-D), tile_dim2=0
  g1[5] = (int)nbytes;                               // tensor_dim0_stride lo32
  g1[6] = 0; g1[7] = 0;
  v4i gz = {0,0,0,0};
#if __clang_major__ >= 23
  v8i gz8 = {0,0,0,0,0,0,0,0};
  __builtin_amdgcn_tensor_load_to_lds(g0, g1, gz, gz, gz8, 0);
#else
  __builtin_amdgcn_tensor_load_to_lds(g0, g1, gz, gz, 0);
#endif
#else
  (void)lds_off; (void)ga; (void)nbytes;
#endif
}

// ---------------- conversion / transposition kernels ----------------

__global__ void cvt_bf16(const float* __restrict__ src, __bf16* __restrict__ dst, int n){
  int i = blockIdx.x*blockDim.x + threadIdx.x;
  if(i < n) dst[i] = (__bf16)src[i];
}

// [nmat][R][C] f32 -> [nmat][C][R] bf16   (n = nmat*R*C)
__global__ void cvt_t_bf16(const float* __restrict__ src, __bf16* __restrict__ dst,
                           int R, int C, int n){
  int i = blockIdx.x*blockDim.x + threadIdx.x;
  if(i >= n) return;
  int m = i / (R*C); int rem = i - m*R*C; int r = rem / C; int c = rem - r*C;
  dst[((size_t)m*C + c)*R + r] = (__bf16)src[i];
}

// ---------------- fused QKV projection ----------------
// grid (B*H, S/64), block 128 (4 waves); wave: 16 s-rows x 64 e-cols x {Q,K,V}
__global__ void __launch_bounds__(128) qkv_proj(
    const __bf16* __restrict__ xb,
    const __bf16* __restrict__ WqT, const __bf16* __restrict__ WkT, const __bf16* __restrict__ WvT,
    const float* __restrict__ bq, const float* __restrict__ bk, const float* __restrict__ bv,
    __bf16* __restrict__ Qs, __bf16* __restrict__ Kb, __bf16* __restrict__ Vt, float invscale)
{
  const int lane = threadIdx.x & 31;
  const int wave = threadIdx.x >> 5;
  const int bh = blockIdx.x; const int b = bh / H_; const int h = bh - b*H_;
  const int s0 = blockIdx.y*64 + wave*16;

  const __bf16* xrow = xb + ((size_t)b*S_ + s0)*D_;
  const __bf16* wq = WqT + (size_t)h*E_*D_;
  const __bf16* wk = WkT + (size_t)h*E_*D_;
  const __bf16* wv = WvT + (size_t)h*E_*D_;

  v8f aq[4], ak[4], av[4];
#pragma unroll
  for(int j=0;j<4;++j){
    float q0v = bq[h*E_ + j*16 + (lane&15)];
    float k0v = bk[h*E_ + j*16 + (lane&15)];
    float v0v = bv[h*E_ + j*16 + (lane&15)];
#pragma unroll
    for(int r=0;r<8;++r){ aq[j][r]=q0v; ak[j][r]=k0v; av[j][r]=v0v; }
  }

  for(int d=0; d<D_; d+=32){
    v16bf a = load_tile16x32(xrow + d, D_, lane);
#pragma unroll
    for(int j=0;j<4;++j){
      v16bf tq = load_tile16x32(wq + (size_t)(j*16)*D_ + d, D_, lane);
      aq[j] = wmma_bf16(a, tq, aq[j]);
      v16bf tk = load_tile16x32(wk + (size_t)(j*16)*D_ + d, D_, lane);
      ak[j] = wmma_bf16(a, tk, ak[j]);
      v16bf tv = load_tile16x32(wv + (size_t)(j*16)*D_ + d, D_, lane);
      av[j] = wmma_bf16(a, tv, av[j]);
    }
  }

  const size_t qk_base = ((size_t)bh*S_ + s0)*E_;
  const size_t vt_base = (size_t)bh*E_*S_;
#pragma unroll
  for(int j=0;j<4;++j){
    int e = j*16 + (lane&15);
#pragma unroll
    for(int r=0;r<8;++r){
      int m = r + ((lane & 16) ? 8 : 0);
      Qs[qk_base + (size_t)m*E_ + e] = (__bf16)(aq[j][r]*invscale); // fold 1/sqrt(S)
      Kb[qk_base + (size_t)m*E_ + e] = (__bf16)(ak[j][r]);
      Vt[vt_base + (size_t)e*S_ + (s0+m)] = (__bf16)(av[j][r]);    // store V transposed
    }
  }
}

// ---------------- streaming (flash) attention ----------------
// grid (B*H, S/64), block 128 (4 waves, 64 queries of one (b,h)).
// Double-buffered LDS staging shared by all 4 waves:
//   K tile (4KB contiguous)  -> Tensor Data Mover, 1 instr from wave 0 (TENSORcnt)
//   V^T tile (strided rows)  -> global_load_async_to_lds_b128      (ASYNCcnt)
__global__ void __launch_bounds__(128) attention(
    const __bf16* __restrict__ Qs, const __bf16* __restrict__ Kb, const __bf16* __restrict__ Vt,
    const unsigned char* __restrict__ mask, __bf16* __restrict__ concat)
{
  __shared__ __align__(16) __bf16 sK[2][32*64];   // [key][e]
  __shared__ __align__(16) __bf16 sV[2][64*32];   // [e][key]
  __shared__ __align__(16) __bf16 sP[4][16*32];   // per-wave P staging

  const int tid  = threadIdx.x;
  const int lane = threadIdx.x & 31;
  const int wave = threadIdx.x >> 5;
  const int bh = blockIdx.x; const int b = bh / H_; const int h = bh - b*H_;
  const int q0 = blockIdx.y*64 + wave*16;

  const __bf16* qbase = Qs + ((size_t)bh*S_ + q0)*E_;
  const __bf16* kbase = Kb + (size_t)bh*S_*E_;
  const __bf16* vbase = Vt + (size_t)bh*E_*S_;
  const unsigned char* mrow = mask + (size_t)b*S_;

  v16bf qa0 = load_tile16x32(qbase,      E_, lane);  // e = 0..31
  v16bf qa1 = load_tile16x32(qbase + 32, E_, lane);  // e = 32..63

  v8f acc[4]; float mrun[8], lrun[8];
#pragma unroll
  for(int j=0;j<4;++j){ v8f z = {}; acc[j] = z; }
#pragma unroll
  for(int r=0;r<8;++r){ mrun[r] = -3.0e38f; lrun[r] = 0.f; }

  // Stage one 32-key tile pair into buffer `buf`.
  auto stage_kv = [&](int t, int buf){
    // K: 32 keys x 64 e bf16, 4096B contiguous -> one TDM descriptor (wave 0 only;
    // TDM ignores EXEC, single issue avoids 4x duplicate DMA).
#if HAVE_TDM
    if(wave == 0){
      tdm_load_1d((unsigned)(uintptr_t)(const void*)&sK[buf][0],
                  (unsigned long long)(uintptr_t)(const void*)(kbase + (size_t)t*E_),
                  32*64*2);
    }
#else
    {
      const char* kgb = (const char*)(kbase + (size_t)t*E_);
      unsigned skoff = (unsigned)(uintptr_t)(const void*)&sK[buf][0];
#pragma unroll
      for(int i=0;i<2;++i){
        int c = tid + 128*i;
        async_copy16(skoff + c*16, kgb + c*16);
      }
    }
#endif
    // V^T: 64 rows (e) x 32 t; 64B per row, 4 chunks per row; 2 async/thread
    const char* vgb = (const char*)vbase;
    unsigned svoff = (unsigned)(uintptr_t)(const void*)&sV[buf][0];
#pragma unroll
    for(int i=0;i<2;++i){
      int c = tid + 128*i;
      int row = c >> 2, off = (c & 3) * 16;
      async_copy16(svoff + row*64 + off,
                   vgb + (size_t)row*(S_*2) + (size_t)t*2 + off);
    }
  };

  stage_kv(0, 0);

  for(int t=0; t<S_; t+=32){
    const int buf = (t >> 5) & 1;
    if(t + 32 < S_){
      stage_kv(t + 32, buf ^ 1);                       // prefetch next tile
#if HAVE_TDM
      asm volatile("s_wait_asynccnt 2" ::: "memory");  // current V tile landed (in-order)
      if(wave == 0) asm volatile("s_wait_tensorcnt 1" ::: "memory"); // current K TDM done
#else
      asm volatile("s_wait_asynccnt 4" ::: "memory");
#endif
    } else {
      asm volatile("s_wait_asynccnt 0" ::: "memory");
#if HAVE_TDM
      if(wave == 0) asm volatile("s_wait_tensorcnt 0" ::: "memory");
#endif
    }
    __syncthreads();

    // scores: two 16x16 tiles over 32 keys, K-dim = E = 64 (2 WMMA each), from LDS
    v16bf kb00 = load_tile16x32(&sK[buf][0],          64, lane);
    v16bf kb01 = load_tile16x32(&sK[buf][32],         64, lane);
    v16bf kb10 = load_tile16x32(&sK[buf][16*64],      64, lane);
    v16bf kb11 = load_tile16x32(&sK[buf][16*64 + 32], 64, lane);
    v8f z = {};
    v8f c0 = wmma_bf16(qa0, kb00, z); c0 = wmma_bf16(qa1, kb01, c0);
    v8f c1 = wmma_bf16(qa0, kb10, z); c1 = wmma_bf16(qa1, kb11, c1);

    // key-side mask (reference: mask broadcast over queries, value -1e9 post-scale)
    bool msk0 = mrow[t +      (lane&15)] != 0;
    bool msk1 = mrow[t + 16 + (lane&15)] != 0;
#pragma unroll
    for(int r=0;r<8;++r){
      if(msk0) c0[r] = -1.0e9f;
      if(msk1) c1[r] = -1.0e9f;
    }

    // online softmax; C layout: lane holds 8 rows (m = r + 8*(lane>=16)), 1 column
#pragma unroll
    for(int r=0;r<8;++r){
      float rmax = fmaxf(c0[r], c1[r]);
#pragma unroll
      for(int o=1;o<16;o<<=1) rmax = fmaxf(rmax, __shfl_xor(rmax, o, 32));
      float mnew  = fmaxf(mrun[r], rmax);
      float alpha = __expf(mrun[r] - mnew);
      float p0 = __expf(c0[r] - mnew);
      float p1 = __expf(c1[r] - mnew);
      float rsum = p0 + p1;
#pragma unroll
      for(int o=1;o<16;o<<=1) rsum += __shfl_xor(rsum, o, 32);
      lrun[r] = lrun[r]*alpha + rsum;
      mrun[r] = mnew;
#pragma unroll
      for(int j=0;j<4;++j) acc[j][r] *= alpha;
      int m = r + ((lane & 16) ? 8 : 0);
      sP[wave][m*32 +      (lane&15)] = (__bf16)p0;   // C-layout -> A-layout via LDS
      sP[wave][m*32 + 16 + (lane&15)] = (__bf16)p1;
    }
    __syncthreads();
    v16bf pa = load_tile16x32(&sP[wave][0], 32, lane);  // P as 16x32 A operand
#pragma unroll
    for(int j=0;j<4;++j){
      v16bf vb = load_tile16x32(&sV[buf][(j*16)*32], 32, lane); // V^T rows = e, from LDS
      acc[j] = wmma_bf16(pa, vb, acc[j]);
    }
    __syncthreads();  // all waves done with buf before it is re-staged
  }

  // normalize and scatter into concat [B,S,H*E] (bf16 for the final GEMM)
#pragma unroll
  for(int j=0;j<4;++j){
    int e = j*16 + (lane&15);
#pragma unroll
    for(int r=0;r<8;++r){
      int m = r + ((lane & 16) ? 8 : 0);
      float v = acc[j][r] / lrun[r];
      concat[((size_t)b*S_ + q0 + m)*(H_*E_) + h*E_ + e] = (__bf16)v;
    }
  }
}

// ---------------- output projection ----------------
// grid (B*S/64, D/64), block 128; wave: 16 rows x 64 cols, K = H*E = 512
__global__ void __launch_bounds__(128) out_proj(
    const __bf16* __restrict__ concat, const __bf16* __restrict__ WoT,
    const float* __restrict__ bo, float* __restrict__ outp)
{
  const int lane = threadIdx.x & 31;
  const int wave = threadIdx.x >> 5;
  const int row0 = blockIdx.x*64 + wave*16;   // over B*S
  const int n0   = blockIdx.y*64;             // over D

  const __bf16* abase = concat + (size_t)row0*(H_*E_);
  v8f acc[4];
#pragma unroll
  for(int j=0;j<4;++j){
    float bias = bo[n0 + j*16 + (lane&15)];
#pragma unroll
    for(int r=0;r<8;++r) acc[j][r] = bias;
  }
  for(int d=0; d<H_*E_; d+=32){
    v16bf a = load_tile16x32(abase + d, H_*E_, lane);
#pragma unroll
    for(int j=0;j<4;++j){
      v16bf bt = load_tile16x32(WoT + (size_t)(n0 + j*16)*(H_*E_) + d, H_*E_, lane);
      acc[j] = wmma_bf16(a, bt, acc[j]);
    }
  }
#pragma unroll
  for(int j=0;j<4;++j){
    int n = n0 + j*16 + (lane&15);
#pragma unroll
    for(int r=0;r<8;++r){
      int m = r + ((lane & 16) ? 8 : 0);
      outp[(size_t)(row0+m)*D_ + n] = acc[j][r];
    }
  }
}

// ---------------- row LayerNorm over D=512 ----------------
__global__ void __launch_bounds__(256) layernorm(
    const float* __restrict__ inp, const float* __restrict__ gamma,
    const float* __restrict__ beta, float* __restrict__ out)
{
  __shared__ float rs[16], rss[16];
  const int row = blockIdx.x;
  const float* p = inp + (size_t)row*D_;
  float a = p[threadIdx.x], c = p[threadIdx.x + 256];
  float s = a + c, ss = a*a + c*c;
#pragma unroll
  for(int o=16;o>0;o>>=1){ s += __shfl_down(s, o, 32); ss += __shfl_down(ss, o, 32); }
  int wid = threadIdx.x >> 5;
  if((threadIdx.x & 31) == 0){ rs[wid] = s; rss[wid] = ss; }
  __syncthreads();
  if(threadIdx.x == 0){
    float S=0.f, SS=0.f;
    for(int i=0;i<8;++i){ S += rs[i]; SS += rss[i]; }
    rs[0] = S; rss[0] = SS;
  }
  __syncthreads();
  float mean = rs[0] * (1.0f/D_);
  float var  = rss[0] * (1.0f/D_) - mean*mean;
  float inv  = rsqrtf(var + 1e-5f);
  out[(size_t)row*D_ + threadIdx.x]       = (a - mean)*inv*gamma[threadIdx.x]       + beta[threadIdx.x];
  out[(size_t)row*D_ + threadIdx.x + 256] = (c - mean)*inv*gamma[threadIdx.x + 256] + beta[threadIdx.x + 256];
}

// ---------------- host launch ----------------
extern "C" void kernel_launch(void* const* d_in, const int* in_sizes, int n_in,
                              void* d_out, int out_size, void* d_ws, size_t ws_size,
                              hipStream_t stream)
{
  (void)in_sizes; (void)n_in; (void)out_size; (void)ws_size;
  const float* x            = (const float*)d_in[0];
  const unsigned char* mask = (const unsigned char*)d_in[1];  // bool array
  const float* Wq = (const float*)d_in[2];
  const float* bq = (const float*)d_in[3];
  const float* Wk = (const float*)d_in[4];
  const float* bk = (const float*)d_in[5];
  const float* Wv = (const float*)d_in[6];
  const float* bv = (const float*)d_in[7];
  const float* Wo = (const float*)d_in[8];
  const float* bo = (const float*)d_in[9];
  const float* gamma = (const float*)d_in[10];
  const float* beta  = (const float*)d_in[11];
  float* out = (float*)d_out;

  char* ws = (char*)d_ws;
  size_t off = 0;
  auto alloc = [&](size_t bytes)->char* {
    char* p = ws + off; off += (bytes + 255) & ~(size_t)255; return p;
  };
  __bf16* xb   = (__bf16*)alloc((size_t)B_*S_*D_*2);
  __bf16* WqT  = (__bf16*)alloc((size_t)H_*E_*D_*2);
  __bf16* WkT  = (__bf16*)alloc((size_t)H_*E_*D_*2);
  __bf16* WvT  = (__bf16*)alloc((size_t)H_*E_*D_*2);
  __bf16* WoT  = (__bf16*)alloc((size_t)D_*H_*E_*2);
  __bf16* Qs   = (__bf16*)alloc((size_t)B_*H_*S_*E_*2);
  __bf16* Kbuf = (__bf16*)alloc((size_t)B_*H_*S_*E_*2);
  __bf16* Vt   = (__bf16*)alloc((size_t)B_*H_*E_*S_*2);
  __bf16* cc   = (__bf16*)alloc((size_t)B_*S_*H_*E_*2);
  float*  outp = (float*)alloc((size_t)B_*S_*D_*4);

  int nx = B_*S_*D_;
  cvt_bf16<<<(nx+255)/256, 256, 0, stream>>>(x, xb, nx);
  int nw = H_*D_*E_;
  cvt_t_bf16<<<(nw+255)/256, 256, 0, stream>>>(Wq, WqT, D_, E_, nw);
  cvt_t_bf16<<<(nw+255)/256, 256, 0, stream>>>(Wk, WkT, D_, E_, nw);
  cvt_t_bf16<<<(nw+255)/256, 256, 0, stream>>>(Wv, WvT, D_, E_, nw);
  int nwo = H_*E_*D_;
  cvt_t_bf16<<<(nwo+255)/256, 256, 0, stream>>>(Wo, WoT, H_*E_, D_, nwo);

  float invscale = 1.0f / sqrtf((float)S_);
  qkv_proj<<<dim3(B_*H_, S_/64), 128, 0, stream>>>(xb, WqT, WkT, WvT, bq, bk, bv,
                                                   Qs, Kbuf, Vt, invscale);
  attention<<<dim3(B_*H_, S_/64), 128, 0, stream>>>(Qs, Kbuf, Vt, mask, cc);
  out_proj<<<dim3(B_*S_/64, D_/64), 128, 0, stream>>>(cc, WoT, bo, outp);
  layernorm<<<B_*S_, 256, 0, stream>>>(outp, gamma, beta, out);
}
